// MultiHeadSelfAttentionWithEdge_68813966016872
// MI455X (gfx1250) — matrix operand order repaired and checked by reading, not compiled, not run on hardware
//
#include <hip/hip_runtime.h>
#include <hip/hip_bf16.h>
#include <math.h>

typedef __attribute__((ext_vector_type(16))) _Float16 v16h;
typedef __attribute__((ext_vector_type(8)))  float    v8f;

#define Bv  2
#define Nv  512
#define Dv  128
#define Hv  8
#define DKv 16

static constexpr int QKV_ELEMS = Bv * Hv * Nv * DKv; // 131072 floats per tensor

// ---------------------------------------------------------------------------
// Kernel 1: QKV projection. out = n @ W_qkv, scattered to [B,H,N,DK] per Q/K/V.
// Q is pre-scaled by DK^-0.5 = 0.25.
// ---------------------------------------------------------------------------
__global__ __launch_bounds__(256) void qkv_prep(const float* __restrict__ nmat,
                                                const float* __restrict__ Wqkv,
                                                float* __restrict__ ws) {
  int gid = blockIdx.x * 256 + threadIdx.x;
  const int total = Bv * Nv * 3 * Dv;
  if (gid >= total) return;
  int row = gid / (3 * Dv);   // b*N + n
  int col = gid % (3 * Dv);   // sel*128 + h*16 + dk
  const float* nrow = nmat + (size_t)row * Dv;
  float acc = 0.f;
#pragma unroll 8
  for (int d = 0; d < Dv; ++d) acc = fmaf(nrow[d], Wqkv[d * (3 * Dv) + col], acc);
  int sel = col / Dv;
  int hd  = col % Dv;
  int h = hd / DKv, dk = hd % DKv;
  int b = row / Nv, nn = row % Nv;
  if (sel == 0) acc *= 0.25f;  // DK^-0.5
  float* dst = ws + (size_t)sel * QKV_ELEMS;
  dst[(((size_t)b * Hv + h) * Nv + nn) * DKv + dk] = acc;
}

// ---------------------------------------------------------------------------
// Kernel 2: fused edge einsums + attention + outputs. One block per (b, nq).
// 256 threads = 8 waves; each wave owns 4 m-tiles of 16 rows.
// ---------------------------------------------------------------------------
__global__ __launch_bounds__(256) void edge_attn_main(
    const float* __restrict__ e,
    const float* __restrict__ We,   // [D,H]
    const float* __restrict__ Wg,   // [D,H]
    const float* __restrict__ On,   // [D,D]
    const float* __restrict__ Oe,   // [H,D]
    const float* __restrict__ ws,
    float* __restrict__ n_out,
    float* __restrict__ e_out)
{
  __shared__ __align__(16) float Qs[Hv * DKv];      // 8 heads x 16, pre-scaled
  __shared__ __align__(16) float OeS[Hv * Dv];      // O_e staged in LDS
  __shared__ __align__(16) float sS[8][16 * 8];     // per-wave _E tile  [m][h]
  __shared__ __align__(16) float pS[8][16 * 8];     // per-wave softmax p [m][h]
  __shared__ __align__(16) float scS[8][8];         // per-wave rescale per head
  __shared__ __align__(16) float accS[8][Hv * DKv]; // per-wave online A@V acc
  __shared__ float mxS[8][8], lS[8][8], gS[8][8];
  __shared__ __align__(16) float vS[Dv];

  const int tid  = threadIdx.x;
  const int wave = tid >> 5;
  const int lane = tid & 31;
  const int j    = lane & 15;  // WMMA C column (0..7 = E head, 8..15 = G head)
  const int half = lane >> 4;
  const int bn = blockIdx.x;
  const int b  = bn >> 9;
  const int nq = bn & (Nv - 1);

  const float* Qw = ws;
  const float* Kw = ws + QKV_ELEMS;
  const float* Vw = ws + 2 * QKV_ELEMS;

  if (tid < 128) {
    int h = tid >> 4, dk = tid & 15;
    Qs[tid] = Qw[(((size_t)b * Hv + h) * Nv + nq) * DKv + dk];
  }
  for (int i = tid; i < Hv * Dv; i += 256) OeS[i] = Oe[i];
  for (int i = lane; i < Hv * DKv; i += 32) accS[wave][i] = 0.f;

  // B-matrix: [W_e | W_g] packed 128x16, WMMA 16-bit B layout:
  // lane column = lane%16, K = chunk*32 + (el/8)*16 + (lane/16)*8 + el%8.
  v16h bmat[4];
  {
    const float* wcol = (j < 8) ? (We + j) : (Wg + (j - 8));
#pragma unroll
    for (int c = 0; c < 4; ++c)
#pragma unroll
      for (int el = 0; el < 16; ++el) {
        int d = c * 32 + (el >> 3) * 16 + half * 8 + (el & 7);
        bmat[c][el] = (_Float16)wcol[d * Hv];
      }
  }
  __syncthreads();

  float mx = -INFINITY, l = 0.f, gl = 0.f;
  const int mrow = lane & 15; // WMMA A row

  for (int it = 0; it < 4; ++it) {
    const int mt = wave + it * 8; // unique tile per (wave, it)
    const int m0 = mt * 16;

    // ---- Phase 1: load e tile in WMMA A layout, f32 -> f16 ----
    // A element el of chunk c: K = c*32 + (el/8)*16 + half*8 + el%8
    v16h a[4];
    const float* erow = e + ((((size_t)b * Nv + nq) * Nv) + (m0 + mrow)) * Dv;
#pragma unroll
    for (int c = 0; c < 4; ++c) {
      const float* p0 = erow + c * 32 + half * 8;
      float tmp[16];
      *(float4*)(tmp + 0)  = *(const float4*)(p0 + 0);
      *(float4*)(tmp + 4)  = *(const float4*)(p0 + 4);
      *(float4*)(tmp + 8)  = *(const float4*)(p0 + 16);
      *(float4*)(tmp + 12) = *(const float4*)(p0 + 20);
#pragma unroll
      for (int el = 0; el < 16; ++el) a[c][el] = (_Float16)tmp[el];
    }

    // ---- Phase 2: E/G einsum on the matrix pipe (K=128 = 4 x 32) ----
    v8f Cacc = {};
#pragma unroll
    for (int c = 0; c < 4; ++c)
      Cacc = __builtin_amdgcn_wmma_f32_16x16x32_f16(
          false, a[c], false, bmat[c], (short)0, Cacc, false, false);
    // C layout: vgpr r, this lane -> (m = half*8 + r, col = j)

    // ---- Phase 3: scores / gate / online softmax bookkeeping ----
    if (j >= 8) {
      // gate columns: sigmoid-sum for dynamic centrality
#pragma unroll
      for (int r = 0; r < 8; ++r) gl += 1.f / (1.f + expf(-Cacc[r]));
    } else {
      float s[8], p[8];
      const float* qh = Qs + j * DKv;
#pragma unroll
      for (int r = 0; r < 8; ++r) {
        int mg = m0 + half * 8 + r;
        const float4* krow =
            (const float4*)(Kw + ((((size_t)b * Hv + j) * Nv) + mg) * DKv);
        float qk = 0.f;
#pragma unroll
        for (int q4 = 0; q4 < 4; ++q4) {
          float4 kv = krow[q4];
          qk = fmaf(qh[q4 * 4 + 0], kv.x, qk);
          qk = fmaf(qh[q4 * 4 + 1], kv.y, qk);
          qk = fmaf(qh[q4 * 4 + 2], kv.z, qk);
          qk = fmaf(qh[q4 * 4 + 3], kv.w, qk);
        }
        qk = fminf(fmaxf(qk, -5.f), 5.f);   // clamp BEFORE adding E
        s[r] = qk + Cacc[r];
      }
      float tmax = s[0];
#pragma unroll
      for (int r = 1; r < 8; ++r) tmax = fmaxf(tmax, s[r]);
      tmax = fmaxf(tmax, __shfl_xor(tmax, 16));
      float nmx = fmaxf(mx, tmax);
      float sc = expf(mx - nmx);
      float psum = 0.f;
#pragma unroll
      for (int r = 0; r < 8; ++r) { p[r] = expf(s[r] - nmx); psum += p[r]; }
      psum += __shfl_xor(psum, 16);
      l = l * sc + psum;
      mx = nmx;
#pragma unroll
      for (int r = 0; r < 8; ++r) {
        int mm = half * 8 + r;
        sS[wave][mm * 8 + j] = s[r];
        pS[wave][mm * 8 + j] = p[r];
      }
      if (half == 0) scS[wave][j] = sc;
    }
    __syncthreads();

    // ---- Phase 4a: e_out tile, fully coalesced 512B row stores ----
    float* eo = e_out + ((((size_t)b * Nv + nq) * Nv) + m0) * Dv;
#pragma unroll 4
    for (int mm = 0; mm < 16; ++mm) {
      float4 o = {0.f, 0.f, 0.f, 0.f};
      const float* srow = &sS[wave][mm * 8];
#pragma unroll
      for (int h = 0; h < 8; ++h) {
        float sv = srow[h];
        float4 ov = *(const float4*)(OeS + h * Dv + lane * 4);
        o.x = fmaf(sv, ov.x, o.x);
        o.y = fmaf(sv, ov.y, o.y);
        o.z = fmaf(sv, ov.z, o.z);
        o.w = fmaf(sv, ov.w, o.w);
      }
      *(float4*)(eo + mm * Dv + lane * 4) = o;
    }

    // ---- Phase 4b: online A@V accumulator update (lane owns h = lane%8,
    //                dk block = (lane/8)*4) ----
    {
      int h = lane & 7, dkb = (lane >> 3) * 4;
      float sc = scS[wave][h];
      float* ap = &accS[wave][h * DKv + dkb];
      float4 acc;
      acc.x = ap[0] * sc; acc.y = ap[1] * sc;
      acc.z = ap[2] * sc; acc.w = ap[3] * sc;
#pragma unroll 4
      for (int mm = 0; mm < 16; ++mm) {
        float pv = pS[wave][mm * 8 + h];
        float4 vv = *(const float4*)(
            Vw + ((((size_t)b * Hv + h) * Nv) + (m0 + mm)) * DKv + dkb);
        acc.x = fmaf(pv, vv.x, acc.x);
        acc.y = fmaf(pv, vv.y, acc.y);
        acc.z = fmaf(pv, vv.z, acc.z);
        acc.w = fmaf(pv, vv.w, acc.w);
      }
      ap[0] = acc.x; ap[1] = acc.y; ap[2] = acc.z; ap[3] = acc.w;
    }
    __syncthreads();
  }

  // ---- Cross-wave merge + outputs ----
  if (j < 8 && half == 0) { mxS[wave][j] = mx; lS[wave][j] = l; }
  if (j >= 8) {
    gl += __shfl_xor(gl, 16);
    if (half == 0) gS[wave][j - 8] = gl;
  }
  __syncthreads();

  if (tid < 128) {
    int h = tid >> 4, dk = tid & 15;
    float M = -INFINITY;
#pragma unroll
    for (int w = 0; w < 8; ++w) M = fmaxf(M, mxS[w][h]);
    float L = 0.f, A = 0.f, g = 0.f;
#pragma unroll
    for (int w = 0; w < 8; ++w) {
      float f = expf(mxS[w][h] - M);
      L += lS[w][h] * f;
      A += accS[w][h * DKv + dk] * f;
      g += gS[w][h];
    }
    vS[tid] = (A / L) * log1pf(g);   // dyn_cent rescale; vS index = h*16+dk
  }
  __syncthreads();

  if (tid < 128) {
    int d = tid;
    float acc = 0.f;
#pragma unroll 8
    for (int k = 0; k < 128; ++k) acc = fmaf(vS[k], On[k * Dv + d], acc);
    n_out[((size_t)b * Nv + nq) * Dv + d] = acc;
  }
}

// ---------------------------------------------------------------------------
extern "C" void kernel_launch(void* const* d_in, const int* in_sizes, int n_in,
                              void* d_out, int out_size, void* d_ws, size_t ws_size,
                              hipStream_t stream) {
  (void)in_sizes; (void)n_in; (void)out_size; (void)ws_size;
  const float* nmat = (const float*)d_in[0];
  const float* e    = (const float*)d_in[1];
  const float* Wqkv = (const float*)d_in[2];
  const float* On   = (const float*)d_in[3];
  const float* Wg   = (const float*)d_in[4];
  const float* We   = (const float*)d_in[5];
  const float* Oe   = (const float*)d_in[6];

  float* out   = (float*)d_out;
  float* n_out = out;                              // [B,N,D]
  float* e_out = out + (size_t)Bv * Nv * Dv;       // [B,N,N,D]
  float* ws    = (float*)d_ws;                     // Q | K | V, 1.5 MB

  const int total_qkv = Bv * Nv * 3 * Dv;
  qkv_prep<<<(total_qkv + 255) / 256, 256, 0, stream>>>(nmat, Wqkv, ws);
  edge_attn_main<<<Bv * Nv, 256, 0, stream>>>(e, We, Wg, On, Oe, ws, n_out, e_out);
}
